// LowRankElmanCell_2396591751746
// MI455X (gfx1250) — matrix operand order repaired and checked by reading, not compiled, not run on hardware
//
#include <hip/hip_runtime.h>
#include <hip/hip_bf16.h>
#include <cstdint>

#define DIMSZ  2048
#define RANKSZ 128
#define TSTEPS 1024
#define BATCH  16
#define TBROWS (TSTEPS * BATCH)   // 16384 rows for the batched GEMMs
#define BD     (BATCH * DIMSZ)    // 32768 parallel scan lanes

typedef __attribute__((ext_vector_type(16))) _Float16 v16h;
typedef __attribute__((ext_vector_type(8)))  _Float16 v8h;
typedef __attribute__((ext_vector_type(8)))  float    v8f;

// Pre-RA scheduling pin: group VMEM-reads then WMMAs so all tile loads of a
// k-step are in flight before the first WMMA consumes any of them.
#if defined(__has_builtin)
#if __has_builtin(__builtin_amdgcn_sched_group_barrier)
#define SCHED_VMEM_THEN_WMMA(nvm, nmm)                         \
  do {                                                         \
    __builtin_amdgcn_sched_group_barrier(0x020, (nvm), 0);     \
    __builtin_amdgcn_sched_group_barrier(0x008, (nmm), 0);     \
  } while (0)
#endif
#endif
#ifndef SCHED_VMEM_THEN_WMMA
#define SCHED_VMEM_THEN_WMMA(nvm, nmm) do {} while (0)
#endif

// ---- WMMA operand loaders (wave32, contiguous K = 16*(lane/16)+s for both A and B) -----------
// A (16x32 f16): caller passes &Arow[k + koff], koff = (lane<16)?0:16.
// Lane elements 0..15 <- base[0..15] (contiguous K). Two b128 loads.
static __device__ __forceinline__ v16h load_a_h(const _Float16* __restrict__ base) {
  const v8h lo = *(const v8h*)(base);
  const v8h hi = *(const v8h*)(base + 8);
  v16h a;
#pragma unroll
  for (int j = 0; j < 8; ++j) { a[j] = lo[j]; a[j + 8] = hi[j]; }
  return a;
}
// B (32x16 f16): caller passes &Brow_col[k + kh], kh = (lane<16)?0:16.
// 16 contiguous K halves for this lane's column. Two b128 loads.
static __device__ __forceinline__ v16h load_b_h(const _Float16* __restrict__ base) {
  const v8h lo = *(const v8h*)(base);
  const v8h hi = *(const v8h*)(base + 8);
  v16h b;
#pragma unroll
  for (int j = 0; j < 8; ++j) { b[j] = lo[j]; b[j + 8] = hi[j]; }
  return b;
}

// ---- f32 -> f16 conversion (n must be a multiple of 8) --------------------
__global__ __launch_bounds__(256) void lre_cvt_f16(
    const float* __restrict__ in, _Float16* __restrict__ out, int n) {
  const int i = (blockIdx.x * 256 + threadIdx.x) * 8;
  if (i + 8 <= n) {
#pragma unroll
    for (int j = 0; j < 8; ++j) out[i + j] = (_Float16)in[i + j];
  }
}

// ---- Stage 1: tmp_x = x*Vx^T, tmp_d = x*Vd^T  ([TB,2048]->[TB,128], f16) --
// Each wave: 16 rows x 64 rank cols (4 r-tiles x {x,d} = 8 accs). Per k-step:
// 18 VMEM reads into distinct buffers, then 8 back-to-back WMMAs (enforced by
// sched_group_barrier) -> one exposed memory round-trip per 8 WMMAs.
__global__ __launch_bounds__(128) void lre_proj_stage1(
    const _Float16* __restrict__ xh, const _Float16* __restrict__ Vxh,
    const _Float16* __restrict__ Vdh,
    _Float16* __restrict__ tx, _Float16* __restrict__ td)
{
  const int lane   = threadIdx.x & 31;
  const int wave   = threadIdx.x >> 5;
  const int tile   = blockIdx.x * 4 + wave;      // (TB/16) * 2 tiles
  const int rhalf  = tile & 1;                   // which 64 rank columns
  const int tbBase = (tile >> 1) * 16;
  const int rHalfBase = rhalf * 64;
  const int row    = lane & 15;
  const int koff   = (lane < 16) ? 0 : 16;
  const int kh     = (lane < 16) ? 0 : 16;

  const v8f zero = {0.f, 0.f, 0.f, 0.f, 0.f, 0.f, 0.f, 0.f};
  v8f accx[4], accd[4];
#pragma unroll
  for (int rt = 0; rt < 4; ++rt) { accx[rt] = zero; accd[rt] = zero; }

  const _Float16* arow  = xh  + (size_t)(tbBase + row) * DIMSZ + koff;
  const _Float16* bxrow = Vxh + (size_t)(rHalfBase + row) * DIMSZ + kh;
  const _Float16* bdrow = Vdh + (size_t)(rHalfBase + row) * DIMSZ + kh;

  for (int k = 0; k < DIMSZ; k += 32) {
    // ---- load phase: A + 8 B tiles (18 b128 loads) ----
    const v16h a = load_a_h(arow + k);
    v16h bx[4], bd[4];
#pragma unroll
    for (int rt = 0; rt < 4; ++rt) {
      bx[rt] = load_b_h(bxrow + (size_t)(rt * 16) * DIMSZ + k);
      bd[rt] = load_b_h(bdrow + (size_t)(rt * 16) * DIMSZ + k);
    }
    // ---- compute phase: 8 consecutive WMMAs (no D->A/B hazards) ----
#pragma unroll
    for (int rt = 0; rt < 4; ++rt) {
      accx[rt] = __builtin_amdgcn_wmma_f32_16x16x32_f16(
          false, a, false, bx[rt], (short)0, accx[rt], false, false);
      accd[rt] = __builtin_amdgcn_wmma_f32_16x16x32_f16(
          false, a, false, bd[rt], (short)0, accd[rt], false, false);
    }
    SCHED_VMEM_THEN_WMMA(18, 8);
  }
  const int mb = (lane < 16) ? 0 : 8;  // C/D layout: VGPR i -> row mb+i, col=row
#pragma unroll
  for (int rt = 0; rt < 4; ++rt) {
    const int cidx = rHalfBase + rt * 16 + row;
#pragma unroll
    for (int i = 0; i < 8; ++i) {
      tx[(size_t)(tbBase + mb + i) * RANKSZ + cidx] = (_Float16)accx[rt][i];
      td[(size_t)(tbBase + mb + i) * RANKSZ + cidx] = (_Float16)accd[rt][i];
    }
  }
}

// ---- Stage 2: wx = tmp_x*Ux^T, wd = tmp_d*Ud^T ([TB,128]->[TB,2048], f16) -
// K=128 -> fully unrolled: 32 b128 loads first, then 8 WMMAs.
__global__ __launch_bounds__(128) void lre_proj_stage2(
    const _Float16* __restrict__ tx, const _Float16* __restrict__ td,
    const _Float16* __restrict__ Uxh, const _Float16* __restrict__ Udh,
    _Float16* __restrict__ wx, _Float16* __restrict__ wd)
{
  const int lane   = threadIdx.x & 31;
  const int wave   = threadIdx.x >> 5;
  const int tile   = blockIdx.x * 4 + wave;      // (TB/16)*(DIM/16) tiles
  const int otile  = tile & (DIMSZ / 16 - 1);
  const int tbBase = (tile >> 7) * 16;           // DIM/16 == 128
  const int oBase  = otile * 16;
  const int row    = lane & 15;
  const int koff   = (lane < 16) ? 0 : 16;
  const int kh     = (lane < 16) ? 0 : 16;

  const _Float16* axrow = tx  + (size_t)(tbBase + row) * RANKSZ + koff;
  const _Float16* adrow = td  + (size_t)(tbBase + row) * RANKSZ + koff;
  const _Float16* bxrow = Uxh + (size_t)(oBase + row) * RANKSZ + kh;
  const _Float16* bdrow = Udh + (size_t)(oBase + row) * RANKSZ + kh;

  v16h Axv[4], Adv[4], Bxv[4], Bdv[4];
#pragma unroll
  for (int kk = 0; kk < 4; ++kk) {
    Axv[kk] = load_a_h(axrow + kk * 32);
    Adv[kk] = load_a_h(adrow + kk * 32);
    Bxv[kk] = load_b_h(bxrow + kk * 32);
    Bdv[kk] = load_b_h(bdrow + kk * 32);
  }
  const v8f zero = {0.f, 0.f, 0.f, 0.f, 0.f, 0.f, 0.f, 0.f};
  v8f ax = zero, ad = zero;
#pragma unroll
  for (int kk = 0; kk < 4; ++kk) {
    ax = __builtin_amdgcn_wmma_f32_16x16x32_f16(false, Axv[kk], false, Bxv[kk], (short)0, ax, false, false);
    ad = __builtin_amdgcn_wmma_f32_16x16x32_f16(false, Adv[kk], false, Bdv[kk], (short)0, ad, false, false);
  }
  SCHED_VMEM_THEN_WMMA(32, 8);
  const int mb = (lane < 16) ? 0 : 8;
#pragma unroll
  for (int i = 0; i < 8; ++i) {
    wx[(size_t)(tbBase + mb + i) * DIMSZ + oBase + row] = (_Float16)ax[i];
    wd[(size_t)(tbBase + mb + i) * DIMSZ + oBase + row] = (_Float16)ad[i];
  }
}

// ---- Sequential scan: one thread per (b,d), register double-buffered ------
__global__ __launch_bounds__(256) void lre_scan(
    const float* __restrict__ x,  const float* __restrict__ h0,
    const _Float16* __restrict__ wx, const _Float16* __restrict__ wd,
    const float* __restrict__ r_h, const float* __restrict__ r_delta,
    const float* __restrict__ b,   const float* __restrict__ b_delta,
    const float* __restrict__ b_gate, float* __restrict__ out)
{
  const int idx = blockIdx.x * blockDim.x + threadIdx.x;  // [0, BD)
  const int dd  = idx & (DIMSZ - 1);
  float h = h0[idx];
  const float rh  = r_h[dd],  rd = r_delta[dd];
  const float bb  = b[dd],    bdl = b_delta[dd], bg = b_gate[dd];
  float* __restrict__ outp = out;                        // [T,B,D]
  float* __restrict__ hp   = out + (size_t)TSTEPS * BD;  // [T+1,B,D]
  hp[idx] = h;                                           // h[0] = h0

  size_t off = idx;
  float xv = x[off];
  float wxv = (float)wx[off], wdv = (float)wd[off];      // prefetch t=0
  for (int t = 0; t < TSTEPS; ++t) {
    float xn = 0.f, wxn = 0.f, wdn = 0.f;                // issue t+1 loads now
    if (t + 1 < TSTEPS) {
      const size_t o2 = off + BD;
      xn = x[o2]; wxn = (float)wx[o2]; wdn = (float)wd[o2];
    }
    const float cand  = tanhf(fmaf(rh, h, wxv + bb));
    const float zdel  = fmaf(rd, h, wdv + bdl);
    const float delta = 1.0f / (1.0f + __expf(-zdel));
    h = fmaf(delta, cand - h, h);                        // (1-d)*h + d*cand
    const float g  = h + xv + bg;
    const float sg = 1.0f / (1.0f + __expf(-g));
    outp[off]     = h * (g * sg);                        // silu gate
    hp[off + BD]  = h;                                   // h[t+1]
    off += BD;
    xv = xn; wxv = wxn; wdv = wdn;
  }
}

extern "C" void kernel_launch(void* const* d_in, const int* in_sizes, int n_in,
                              void* d_out, int out_size, void* d_ws, size_t ws_size,
                              hipStream_t stream) {
  (void)in_sizes; (void)n_in; (void)out_size; (void)ws_size;
  const float* x   = (const float*)d_in[0];
  const float* h0  = (const float*)d_in[1];
  const float* Ux  = (const float*)d_in[2];
  const float* Vx  = (const float*)d_in[3];
  const float* Ud  = (const float*)d_in[4];
  const float* Vd  = (const float*)d_in[5];
  const float* r_h = (const float*)d_in[6];
  const float* r_d = (const float*)d_in[7];
  const float* b   = (const float*)d_in[8];
  const float* b_d = (const float*)d_in[9];
  const float* b_g = (const float*)d_in[10];
  float* out = (float*)d_out;

  // f16 workspace: xh(64MB) | Vxh,Vdh,Uxh,Udh(0.5MB ea) | tx,td(4MB ea) | wx,wd(64MB ea)
  _Float16* xh  = (_Float16*)d_ws;
  _Float16* Vxh = xh  + (size_t)TBROWS * DIMSZ;
  _Float16* Vdh = Vxh + (size_t)RANKSZ * DIMSZ;
  _Float16* Uxh = Vdh + (size_t)RANKSZ * DIMSZ;
  _Float16* Udh = Uxh + (size_t)DIMSZ * RANKSZ;
  _Float16* tx  = Udh + (size_t)DIMSZ * RANKSZ;
  _Float16* td  = tx  + (size_t)TBROWS * RANKSZ;
  _Float16* wx  = td  + (size_t)TBROWS * RANKSZ;
  _Float16* wd  = wx  + (size_t)TBROWS * DIMSZ;

  const int nx = TBROWS * DIMSZ;        // 33,554,432
  const int nm = RANKSZ * DIMSZ;        // 262,144
  lre_cvt_f16<<<nx / 8 / 256, 256, 0, stream>>>(x,  xh,  nx);
  lre_cvt_f16<<<nm / 8 / 256, 256, 0, stream>>>(Vx, Vxh, nm);
  lre_cvt_f16<<<nm / 8 / 256, 256, 0, stream>>>(Vd, Vdh, nm);
  lre_cvt_f16<<<nm / 8 / 256, 256, 0, stream>>>(Ux, Uxh, nm);
  lre_cvt_f16<<<nm / 8 / 256, 256, 0, stream>>>(Ud, Udh, nm);

  lre_proj_stage1<<<(TBROWS / 16) * 2 / 4, 128, 0, stream>>>(xh, Vxh, Vdh, tx, td);
  lre_proj_stage2<<<(TBROWS / 16) * (DIMSZ / 16) / 4, 128, 0, stream>>>(tx, td, Uxh, Udh, wx, wd);
  lre_scan<<<BD / 256, 256, 0, stream>>>(x, h0, wx, wd, r_h, r_d, b, b_d, b_g, out);
}
